// SparseConvNeXtBlock_2327872274475
// MI455X (gfx1250) — matrix operand order repaired and checked by reading, not compiled, not run on hardware
//
#include <hip/hip_runtime.h>
#include <hip/hip_bf16.h>
#include <math.h>
#include <stdint.h>

// ---------------- CDNA5 WMMA types ----------------
typedef __attribute__((ext_vector_type(16))) __bf16 v16bf;
typedef __attribute__((ext_vector_type(8)))  float  v8f;

struct __attribute__((aligned(16))) U4  { unsigned int x, y, z, w; };
struct __attribute__((aligned(16))) B32 { U4 lo, hi; };
struct __attribute__((aligned(16))) F4  { float x, y, z, w; };

static __device__ __forceinline__ v16bf mk16(const U4& a, const U4& b) {
  B32 t{a, b};
  return __builtin_bit_cast(v16bf, t);
}

// round-to-nearest-even f32 -> bf16
static __device__ __forceinline__ unsigned short f2bf(float f) {
  unsigned int u = __float_as_uint(f);
  u += 0x7FFFu + ((u >> 16) & 1u);
  return (unsigned short)(u >> 16);
}

// Branch-free exact-form GELU: 0.5x(1+erf(x/sqrt2)), erf via Abramowitz-Stegun
// 7.1.26 (|err|<=1.5e-7). Uses v_rcp_f32 / v_exp_f32 trans ops, no exec divergence.
static __device__ __forceinline__ float gelu_fast(float xv) {
  const float z  = 0.70710678118654752f * xv;
  const float az = fabsf(z);
  const float t  = __builtin_amdgcn_rcpf(fmaf(0.3275911f, az, 1.0f));
  float pl = fmaf(t, 1.061405429f, -1.453152027f);
  pl = fmaf(t, pl, 1.421413741f);
  pl = fmaf(t, pl, -0.284496736f);
  pl = fmaf(t, pl, 0.254829592f);
  pl *= t;
  const float e = __builtin_amdgcn_exp2f(az * az * -1.4426950408889634f);
  float er = fmaf(-pl, e, 1.0f);                       // erf(|z|)
  er = __uint_as_float(__float_as_uint(er) | (__float_as_uint(z) & 0x80000000u));
  return 0.5f * xv * (1.0f + er);
}

// ---------------- problem constants ----------------
#define CDIM  256
#define HDIM  1024
#define HWN   3136          // 56*56
#define TM    64            // positions per workgroup tile (49 tiles / image)
#define YS    264           // y-tile row stride (bf16 elems), 528B (16B aligned rows)
#define HS    1032          // h-tile row stride (bf16 elems), 2064B (16B aligned rows)
#define SMEM_BYTES (TM*YS*2 + TM*HS*2 + 512*4 + 128*4 + 64*4)   // 168704

// ---------------- weight prep: f32 (K,N) -> bf16 (N,K) ----------------
__global__ void prep_weights(const float* __restrict__ w1,
                             const float* __restrict__ w2,
                             unsigned short* __restrict__ w1t,
                             unsigned short* __restrict__ w2t) {
  const int idx = blockIdx.x * 256 + threadIdx.x;      // 0 .. 262143 exactly
  {   // w1 (256,1024) -> w1t[n][k], n<1024, k<256
    const int n = idx >> 8, k = idx & 255;
    w1t[idx] = f2bf(w1[k * HDIM + n]);
  }
  {   // w2 (1024,256) -> w2t[c][k], c<256, k<1024
    const int c = idx >> 10, k = idx & 1023;
    w2t[idx] = f2bf(w2[k * CDIM + c]);
  }
}

// ---------------- fused ConvNeXt block ----------------
__global__ void __launch_bounds__(256)
sparse_convnext_fused(const float* __restrict__ x,
                      const int*   __restrict__ mask,
                      const float* __restrict__ dww,
                      const float* __restrict__ dwb,
                      const float* __restrict__ lnw,
                      const float* __restrict__ lnb,
                      const float* __restrict__ b1,
                      const float* __restrict__ b2,
                      const float* __restrict__ gamma,
                      const unsigned short* __restrict__ w1t,
                      const unsigned short* __restrict__ w2t,
                      float* __restrict__ out) {
  extern __shared__ __align__(16) char smem[];
  unsigned short* y_tile = (unsigned short*)smem;                 // TM x YS bf16
  char* hmem = smem + (size_t)TM * YS * 2;
  unsigned short* h_tile = (unsigned short*)hmem;                 // TM x HS bf16
  float* vstage = (float*)hmem;                                   // TM x 256 f32 overlay (pre-GEMM1 only)
  float* red    = (float*)(hmem + (size_t)TM * HS * 2);           // 512 f32
  float* stats  = red + 512;                                      // 128 f32
  float* pmask  = stats + 128;                                    // 64 f32 per-position mask

  const int t   = threadIdx.x;
  const int T   = blockIdx.x;
  const int b   = T / 49;
  const int hw0 = (T - b * 49) * TM;

  // ======== Stage A1: depthwise 7x7 conv + bias, LN partial sums ========
  const int p  = t & 63;
  const int cg = t >> 6;
  const int hw = hw0 + p;
  const int h0 = hw / 56;
  const int w0 = hw - h0 * 56;
  const float mval = (float)mask[b * 49 + (h0 >> 3) * 7 + (w0 >> 3)];
  if (cg == 0) pmask[p] = mval;

  float s = 0.f, ss = 0.f;
  for (int i = 0; i < 64; ++i) {
    const int c = cg * 64 + i;
    const float* xc = x + ((size_t)(b * CDIM + c)) * HWN;
    const float* wc = dww + c * 49;
    float v = dwb[c];
#pragma unroll
    for (int r7 = 0; r7 < 7; ++r7) {
      const int hh = h0 + r7 - 3;
      if ((unsigned)hh < 56u) {
        const float* xr = xc + hh * 56;
#pragma unroll
        for (int c7 = 0; c7 < 7; ++c7) {
          const int ww = w0 + c7 - 3;
          if ((unsigned)ww < 56u) v = fmaf(xr[ww], wc[r7 * 7 + c7], v);
        }
      }
    }
    vstage[p * CDIM + c] = v;
    s += v; ss += v * v;
  }
  red[cg * 64 + p]       = s;
  red[256 + cg * 64 + p] = ss;
  __syncthreads();

  // ======== Stage A2: LN statistics per position ========
  if (t < 64) {
    const float sum = red[t] + red[64 + t] + red[128 + t] + red[192 + t];
    const float sq  = red[256 + t] + red[320 + t] + red[384 + t] + red[448 + t];
    const float mu  = sum * (1.0f / 256.0f);
    const float var = sq * (1.0f / 256.0f) - mu * mu;
    stats[t]      = mu;
    stats[64 + t] = rsqrtf(var + 1e-6f);
  }
  __syncthreads();

  // ======== Stage A3: normalize + mask -> y_tile (bf16, K-contiguous) ========
  {
    const float mu = stats[p], rs = stats[64 + p];
    for (int i = 0; i < 64; ++i) {
      const int c = cg * 64 + i;
      const float v  = vstage[p * CDIM + c];
      const float yv = ((v - mu) * rs * lnw[c] + lnb[c]) * mval;
      y_tile[p * YS + c] = f2bf(yv);
    }
  }
  __syncthreads();

  // ======== wave / lane geometry for WMMA (ISA 7.12.2 layouts) ========
  const int wave = t >> 5;
  const int lane = t & 31;
  const int rb   = wave >> 1;       // row block 0..3 (16 rows each)
  const int cwv  = wave & 1;        // column half
  const int n15  = lane & 15;
  const int khi  = lane >> 4;       // 0/1

  // ======== GEMM1: H = gelu(Y @ W1 + b1), 64x1024 into LDS ========
  // 2-wide N blocking: one A fragment feeds two WMMA accumulator chains.
  const int arow1 = (16 * rb + n15) * YS + khi * 8;
  for (int nt = 0; nt < 16; ++nt) {
    const int nc0 = cwv * 512 + nt * 32 + n15;           // hidden columns nc0, nc0+16
    const int nc1 = nc0 + 16;
    const unsigned short* br0 = w1t + (size_t)nc0 * CDIM + khi * 8;
    const unsigned short* br1 = w1t + (size_t)nc1 * CDIM + khi * 8;
    if (nt < 15) __builtin_prefetch(br0 + 32 * CDIM, 0, 1);
    v8f acc0 = {}, acc1 = {};
#pragma unroll
    for (int kc = 0; kc < 8; ++kc) {
      const int k0 = kc * 32;
      const v16bf A = mk16(*(const U4*)(y_tile + arow1 + k0),
                           *(const U4*)(y_tile + arow1 + k0 + 16));
      const v16bf B0 = mk16(*(const U4*)(br0 + k0), *(const U4*)(br0 + k0 + 16));
      const v16bf B1 = mk16(*(const U4*)(br1 + k0), *(const U4*)(br1 + k0 + 16));
      acc0 = __builtin_amdgcn_wmma_f32_16x16x32_bf16(false, A, false, B0,
                                                     (short)0, acc0, false, false);
      acc1 = __builtin_amdgcn_wmma_f32_16x16x32_bf16(false, A, false, B1,
                                                     (short)0, acc1, false, false);
    }
    const float bb0  = b1[nc0];
    const float bb1  = b1[nc1];
    const int rbase  = 16 * rb + khi * 8;
#pragma unroll
    for (int i = 0; i < 8; ++i) {
      h_tile[(rbase + i) * HS + nc0] = f2bf(gelu_fast(acc0[i] + bb0));
      h_tile[(rbase + i) * HS + nc1] = f2bf(gelu_fast(acc1[i] + bb1));
    }
  }
  __syncthreads();

  // ======== GEMM2: Z = (H @ W2 + b2)*gamma*mask ; out = x + Z ========
  const int arow2 = (16 * rb + n15) * HS + khi * 8;
  const int pl0   = 16 * rb + khi * 8;    // this lane's 8 local positions (fixed for all tiles)
  float mrow[8];
#pragma unroll
  for (int i = 0; i < 8; ++i) mrow[i] = pmask[pl0 + i];

  for (int nt = 0; nt < 4; ++nt) {
    const int nc0 = cwv * 128 + nt * 32 + n15;           // output channels nc0, nc0+16
    const int nc1 = nc0 + 16;
    const unsigned short* br0 = w2t + (size_t)nc0 * HDIM + khi * 8;
    const unsigned short* br1 = w2t + (size_t)nc1 * HDIM + khi * 8;
    if (nt < 3) __builtin_prefetch(br0 + 32 * HDIM, 0, 1);
    v8f acc0 = {}, acc1 = {};
#pragma unroll 4
    for (int kc = 0; kc < 32; ++kc) {
      const int k0 = kc * 32;
      const v16bf A = mk16(*(const U4*)(h_tile + arow2 + k0),
                           *(const U4*)(h_tile + arow2 + k0 + 16));
      const v16bf B0 = mk16(*(const U4*)(br0 + k0), *(const U4*)(br0 + k0 + 16));
      const v16bf B1 = mk16(*(const U4*)(br1 + k0), *(const U4*)(br1 + k0 + 16));
      acc0 = __builtin_amdgcn_wmma_f32_16x16x32_bf16(false, A, false, B0,
                                                     (short)0, acc0, false, false);
      acc1 = __builtin_amdgcn_wmma_f32_16x16x32_bf16(false, A, false, B1,
                                                     (short)0, acc1, false, false);
    }
    const int pbase = hw0 + pl0;          // multiple of 8 -> 32B aligned
#pragma unroll 2
    for (int half = 0; half < 2; ++half) {
      const int   nc  = half ? nc1 : nc0;
      const v8f&  acc = half ? acc1 : acc0;
      const float bb  = b2[nc];
      const float gg  = gamma[nc];
      const size_t goff = ((size_t)(b * CDIM + nc)) * HWN + pbase;
      const F4 x0 = *(const F4*)(x + goff);
      const F4 x1 = *(const F4*)(x + goff + 4);
      F4 o0 = { x0.x + (acc[0] + bb) * gg * mrow[0],
                x0.y + (acc[1] + bb) * gg * mrow[1],
                x0.z + (acc[2] + bb) * gg * mrow[2],
                x0.w + (acc[3] + bb) * gg * mrow[3] };
      F4 o1 = { x1.x + (acc[4] + bb) * gg * mrow[4],
                x1.y + (acc[5] + bb) * gg * mrow[5],
                x1.z + (acc[6] + bb) * gg * mrow[6],
                x1.w + (acc[7] + bb) * gg * mrow[7] };
      *(F4*)(out + goff)     = o0;
      *(F4*)(out + goff + 4) = o1;
    }
  }
}

// ---------------- host launch ----------------
extern "C" void kernel_launch(void* const* d_in, const int* in_sizes, int n_in,
                              void* d_out, int out_size, void* d_ws, size_t ws_size,
                              hipStream_t stream) {
  const float* x     = (const float*)d_in[0];
  const int*   mask  = (const int*)  d_in[1];
  const float* dw_w  = (const float*)d_in[2];
  const float* dw_b  = (const float*)d_in[3];
  const float* ln_w  = (const float*)d_in[4];
  const float* ln_b  = (const float*)d_in[5];
  const float* w1    = (const float*)d_in[6];
  const float* b1    = (const float*)d_in[7];
  const float* w2    = (const float*)d_in[8];
  const float* b2    = (const float*)d_in[9];
  const float* gamma = (const float*)d_in[10];

  unsigned short* w1t = (unsigned short*)d_ws;                  // 1024x256 bf16
  unsigned short* w2t = w1t + (size_t)HDIM * CDIM;              // 256x1024 bf16
  float* out = (float*)d_out;

  // Allow >64KB dynamic LDS (CDNA5: 320KB/WGP). Idempotent, capture-safe.
  hipFuncSetAttribute((const void*)sparse_convnext_fused,
                      hipFuncAttributeMaxDynamicSharedMemorySize, SMEM_BYTES);

  prep_weights<<<(HDIM * CDIM) / 256, 256, 0, stream>>>(w1, w2, w1t, w2t);

  const int tiles = 32 * (HWN / TM);   // 32 images * 49 tiles = 1568
  sparse_convnext_fused<<<tiles, 256, SMEM_BYTES, stream>>>(
      x, mask, dw_w, dw_b, ln_w, ln_b, b1, b2, gamma, w1t, w2t, out);
}
